// Group_SA_19550691131523
// MI455X (gfx1250) — matrix operand order repaired and checked by reading, not compiled, not run on hardware
//
#include <hip/hip_runtime.h>
#include <hip/hip_bf16.h>

// ---------------------------------------------------------------------------
// Grouped self-attention + cross-attention block for MI455X (gfx1250).
// All matmuls run through v_wmma_f32_16x16x32_f16 (f16 operands, f32 accum).
// One generic batched GEMM kernel (128x128 tile, 8 waves, 8 WMMA/wave/K-step)
// serves: QK proj (block-diag weight, q==k computed once), V proj (channel-
// major f16 out), energy Q^T K, attention-apply (transposed-P A operand per
// the reference's einsum, via templated TRANSA path), FFN fc1/fc2, and all
// cross-layer GEMMs.  Tile loads use gfx1250 async global->LDS copies
// (global_load_async_to_lds_b128) in a double-buffered pipeline: tile i+1 is
// in flight while tile i is consumed; s_wait_asynccnt <= N only waits for the
// previous tile.  Epilogue variants (bias/resid/relu/out-format) are template
// parameters so each instantiation is branchless straight-line code.
// Scratch (~565 MB + 40KB LDS/block) lives in d_ws with buffer overlays.
// ---------------------------------------------------------------------------

#define USE_ASYNC_COPY 1

typedef _Float16 half_t;
typedef __attribute__((ext_vector_type(16))) _Float16 v16h;
typedef __attribute__((ext_vector_type(4)))  _Float16 v4h;
typedef __attribute__((ext_vector_type(8)))  float    v8f;

constexpr int B_  = 8;     // batch
constexpr int C_  = 256;   // channels
constexpr int DH_ = 64;    // head dim
constexpr int GP_ = 4;     // groups / partitions
constexpr int NF_ = 4096;  // full sequence
constexpr int N_  = 1024;  // per-partition sequence
constexpr int BS_ = 32;    // GP_*B_ sub-batches
constexpr int C4_ = 1024;  // FFN hidden

// ---------------------------------------------------------------------------
// Generic batched WMMA GEMM:
//   Out[bt][t][o] = act( scale * sum_k A[bt][t][k] * Bw[bt%bmod][o][k]
//                        + bias[o] + resid[bt][t][o] )
// TRANSA!=0 reads A as Asrc[k][t] (row stride T), transposing through LDS.
// OMODE: 0 = f32 out, 1 = f16 out, 2 = f16 transposed out ([o][t]).
// T, O multiples of 128; K multiple of 32.
// ---------------------------------------------------------------------------
template <int TRANSA, int BIAS, int RESID, int RELU, int OMODE>
__global__ void __launch_bounds__(256)
gemm_wmma(const half_t* __restrict__ A, const half_t* __restrict__ Bw,
          const float* __restrict__ bias, const float* __restrict__ resid,
          float* __restrict__ outF, half_t* __restrict__ outH,
          half_t* __restrict__ outHT,
          int T, int O, int K,
          long sA, long sB, int bmod,
          long sOF, long sOH, long sOHT, long sR,
          float scale)
{
    const int bt   = blockIdx.z;
    const int m0   = blockIdx.x * 128;  // token tile
    const int n0   = blockIdx.y * 128;  // output-col tile
    const int tid  = threadIdx.x;
    const int wave = tid >> 5;
    const int lane = tid & 31;
    const int l15  = lane & 15;
    const int lh   = lane >> 4;
    const int wr   = wave >> 1;         // wave row group: 32 rows
    const int wc   = wave & 1;          // wave col group: 64 cols

    const half_t* Ab = A  + (long)bt * sA;
    const half_t* Bb = Bw + (long)(bt % bmod) * sB;

    // double-buffered tiles: [stage][row][k], 80B row stride (16B aligned)
    __shared__ half_t As[2][128][40];
    __shared__ half_t Bs[2][128][40];

    v8f acc[2][4];
#pragma unroll
    for (int i = 0; i < 2; ++i)
#pragma unroll
        for (int j = 0; j < 4; ++j) acc[i][j] = v8f{};

    const int r    = tid >> 1;          // 0..127
    const int s    = (tid & 1) * 16;    // halves 0 / 16
    const int kkT  = tid >> 3;          // TRANSA: 0..31
    const int segT = (tid & 7) * 16;    // TRANSA: t-offset within tile

    // wave-relative LDS byte addresses for async global->LDS copies
    const unsigned ldsA[2] = {
        (unsigned)(unsigned long long)(uintptr_t)&As[0][r][s],
        (unsigned)(unsigned long long)(uintptr_t)&As[1][r][s]};
    const unsigned ldsB[2] = {
        (unsigned)(unsigned long long)(uintptr_t)&Bs[0][r][s],
        (unsigned)(unsigned long long)(uintptr_t)&Bs[1][r][s]};

    auto issue_tile = [&](int k0i, int buf) {
        const half_t* gB = Bb + (long)(n0 + r) * K + k0i + s;
        __builtin_prefetch(gB + 64, 0, 1);  // speculative, 2 tiles ahead
#if USE_ASYNC_COPY
        asm volatile(
            "global_load_async_to_lds_b128 %0, %1, off\n\t"
            "global_load_async_to_lds_b128 %0, %1, off offset:16"
            :: "v"(ldsB[buf]), "v"((unsigned long long)(uintptr_t)gB)
            : "memory");
        if (!TRANSA) {
            const half_t* gA = Ab + (long)(m0 + r) * K + k0i + s;
            __builtin_prefetch(gA + 64, 0, 1);
            asm volatile(
                "global_load_async_to_lds_b128 %0, %1, off\n\t"
                "global_load_async_to_lds_b128 %0, %1, off offset:16"
                :: "v"(ldsA[buf]), "v"((unsigned long long)(uintptr_t)gA)
                : "memory");
        }
#else
        {
            const uint4* srcB = (const uint4*)gB;
            uint4 b0 = srcB[0], b1 = srcB[1];
            *(uint4*)&Bs[buf][r][s]     = b0;
            *(uint4*)&Bs[buf][r][s + 8] = b1;
            if (!TRANSA) {
                const uint4* srcA =
                    (const uint4*)(Ab + (long)(m0 + r) * K + k0i + s);
                uint4 a0 = srcA[0], a1 = srcA[1];
                *(uint4*)&As[buf][r][s]     = a0;
                *(uint4*)&As[buf][r][s + 8] = a1;
            }
        }
#endif
    };

    auto wait_prev_tile = [&]() {
#if USE_ASYNC_COPY
        if (TRANSA) asm volatile("s_wait_asynccnt 2" ::: "memory");
        else        asm volatile("s_wait_asynccnt 4" ::: "memory");
#endif
    };

    auto load_aT = [&](int k0i, uint4& t0, uint4& t1) {
        const uint4* srcA = (const uint4*)(Ab + (long)(k0i + kkT) * T + m0 + segT);
        t0 = srcA[0]; t1 = srcA[1];
    };
    auto store_aT = [&](int buf, uint4 t0, uint4 t1) {
        union { uint4 u; half_t h[8]; } u0, u1;
        u0.u = t0; u1.u = t1;
#pragma unroll
        for (int i = 0; i < 8; ++i) {
            As[buf][segT + i][kkT]     = u0.h[i];
            As[buf][segT + 8 + i][kkT] = u1.h[i];
        }
    };

    auto compute = [&](int buf) {
        union HU { v16h v; uint4 u[2]; };
        HU bf[4], af[2];
#pragma unroll
        for (int j = 0; j < 4; ++j) {       // B frag: lane=col, ISA K order
            const half_t* brow = &Bs[buf][wc * 64 + j * 16 + l15][0];
            bf[j].u[0] = *(const uint4*)(brow + lh * 8);
            bf[j].u[1] = *(const uint4*)(brow + 16 + lh * 8);
        }
#pragma unroll
        for (int i = 0; i < 2; ++i) {       // A frag: lane=row, ISA K order
            const half_t* arow = &As[buf][wr * 32 + i * 16 + l15][0];
            af[i].u[0] = *(const uint4*)(arow + lh * 8);
            af[i].u[1] = *(const uint4*)(arow + 16 + lh * 8);
        }
#pragma unroll
        for (int i = 0; i < 2; ++i)
#pragma unroll
            for (int j = 0; j < 4; ++j)
                acc[i][j] = __builtin_amdgcn_wmma_f32_16x16x32_f16(
                    false, af[i].v, false, bf[j].v, (short)0, acc[i][j],
                    false, false);
    };

    // -------------------- double-buffered main loop ---------------------
    const int nsteps = K >> 5;
    uint4 ta0, ta1;
    if (TRANSA) load_aT(0, ta0, ta1);
    issue_tile(0, 0);
    if (TRANSA) store_aT(0, ta0, ta1);

    for (int i = 0; i < nsteps; ++i) {
        const int cur   = i & 1;
        const int nxt   = cur ^ 1;
        const int knext = (i + 1 < nsteps) ? (i + 1) * 32 : 0;  // clamped tail
        if (TRANSA) load_aT(knext, ta0, ta1);
        __syncthreads();             // buf[nxt] readers (tile i-1) are done
        issue_tile(knext, nxt);      // in flight during compute of tile i
        if (TRANSA) store_aT(nxt, ta0, ta1);
        wait_prev_tile();            // tile i resident in buf[cur]
        __syncthreads();
        compute(cur);
    }

    // ---- epilogue: D layout lane=col, VGPR rr -> row rr + 8*h ----------
#pragma unroll
    for (int i = 0; i < 2; ++i) {
#pragma unroll
        for (int j = 0; j < 4; ++j) {
            const int o  = n0 + wc * 64 + j * 16 + l15;
            float bv = 0.0f;
            if (BIAS) bv = bias[o];
#pragma unroll
            for (int rr = 0; rr < 8; ++rr) {
                const int t = m0 + wr * 32 + i * 16 + rr + 8 * lh;
                float vv = acc[i][j][rr] * scale + bv;
                const long idx = (long)t * O + o;
                if (RESID) vv += resid[(long)bt * sR + idx];
                if (RELU)  vv = fmaxf(vv, 0.0f);
                if (OMODE == 0)      outF[(long)bt * sOF + idx] = vv;
                else if (OMODE == 1) outH[(long)bt * sOH + idx] = (half_t)vv;
                else outHT[(long)bt * sOHT + (long)o * T + t] = (half_t)vv;
            }
        }
    }
}

// --------------------------- softmax over 1024 ------------------------------
__global__ void __launch_bounds__(256)
softmax_row(const float* __restrict__ E, half_t* __restrict__ P)
{
    const long row = blockIdx.x;
    const int  tid = threadIdx.x;
    const float4 v = ((const float4*)(E + row * 1024))[tid];
    __shared__ float red[8];

    float mx = fmaxf(fmaxf(v.x, v.y), fmaxf(v.z, v.w));
#pragma unroll
    for (int o = 16; o > 0; o >>= 1) mx = fmaxf(mx, __shfl_xor(mx, o, 32));
    if ((tid & 31) == 0) red[tid >> 5] = mx;
    __syncthreads();
    mx = red[0];
#pragma unroll
    for (int i = 1; i < 8; ++i) mx = fmaxf(mx, red[i]);

    const float e0 = __expf(v.x - mx), e1 = __expf(v.y - mx);
    const float e2 = __expf(v.z - mx), e3 = __expf(v.w - mx);
    float s = e0 + e1 + e2 + e3;
#pragma unroll
    for (int o = 16; o > 0; o >>= 1) s += __shfl_xor(s, o, 32);
    __syncthreads();
    if ((tid & 31) == 0) red[tid >> 5] = s;
    __syncthreads();
    s = red[0];
#pragma unroll
    for (int i = 1; i < 8; ++i) s += red[i];
    const float inv = 1.0f / s;

    v4h h;
    h[0] = (half_t)(e0 * inv); h[1] = (half_t)(e1 * inv);
    h[2] = (half_t)(e2 * inv); h[3] = (half_t)(e3 * inv);
    ((v4h*)(P + row * 1024))[tid] = h;
}

// --------------------------- LayerNorm over C=256 ---------------------------
__global__ void __launch_bounds__(64)
layernorm_tok(const float* __restrict__ X, const float* __restrict__ gam,
              const float* __restrict__ bet, float* __restrict__ outF,
              half_t* __restrict__ outH, int relu)
{
    const long t  = blockIdx.x;
    const int tid = threadIdx.x;
    const float4 v = ((const float4*)(X + t * 256))[tid];
    float s = v.x + v.y + v.z + v.w;
    float q = v.x * v.x + v.y * v.y + v.z * v.z + v.w * v.w;
#pragma unroll
    for (int o = 16; o > 0; o >>= 1) {
        s += __shfl_xor(s, o, 32);
        q += __shfl_xor(q, o, 32);
    }
    __shared__ float rs[2], rq[2];
    if ((tid & 31) == 0) { rs[tid >> 5] = s; rq[tid >> 5] = q; }
    __syncthreads();
    s = rs[0] + rs[1]; q = rq[0] + rq[1];
    const float mean = s * (1.0f / 256.0f);
    const float var  = q * (1.0f / 256.0f) - mean * mean;
    const float inv  = rsqrtf(var + 1e-6f);
    const float4 g4 = ((const float4*)gam)[tid];
    const float4 b4 = ((const float4*)bet)[tid];
    float y0 = (v.x - mean) * inv * g4.x + b4.x;
    float y1 = (v.y - mean) * inv * g4.y + b4.y;
    float y2 = (v.z - mean) * inv * g4.z + b4.z;
    float y3 = (v.w - mean) * inv * g4.w + b4.w;
    if (relu) {
        y0 = fmaxf(y0, 0.f); y1 = fmaxf(y1, 0.f);
        y2 = fmaxf(y2, 0.f); y3 = fmaxf(y3, 0.f);
    }
    ((float4*)(outF + t * 256))[tid] = make_float4(y0, y1, y2, y3);
    if (outH) {
        v4h h;
        h[0] = (half_t)y0; h[1] = (half_t)y1;
        h[2] = (half_t)y2; h[3] = (half_t)y3;
        ((v4h*)(outH + t * 256))[tid] = h;
    }
}

// ------------------- input permute: feat -> token-major gx ------------------
// x[(nn%4)*8+b][nn/4][c] = feat[b][c][nn]   (LDS-tiled transpose)
__global__ void __launch_bounds__(256)
permute_in(const float* __restrict__ feat, float* __restrict__ xt,
           half_t* __restrict__ xh)
{
    __shared__ float tile[32][33];
    const int b   = blockIdx.z;
    const int c0  = blockIdx.y * 32;
    const int nn0 = blockIdx.x * 32;
    const int tx  = threadIdx.x;   // 32
    const int ty  = threadIdx.y;   // 8
    for (int cc = ty; cc < 32; cc += 8)
        tile[cc][tx] = feat[((long)b * C_ + c0 + cc) * NF_ + nn0 + tx];
    __syncthreads();
    for (int jj = ty; jj < 32; jj += 8) {
        const int nn = nn0 + jj;
        const int gp = nn & 3, ni = nn >> 2;
        const long dst = (((long)(gp * B_ + b)) * N_ + ni) * C_ + c0 + tx;
        const float vv = tile[jj][tx];
        xt[dst] = vv;
        xh[dst] = (half_t)vv;
    }
}

// ---------------- SA-out -> cross-layer-in token permutation ----------------
// cx[gp2*8+b][j][:] = z[(nn%4)*8+b][nn/4][:],  nn = gp2*1024 + j
__global__ void __launch_bounds__(64)
permute_tok(const float* __restrict__ z, float* __restrict__ cx,
            half_t* __restrict__ cxh)
{
    const int j   = blockIdx.x;
    const int bs2 = blockIdx.y;
    const int gp2 = bs2 >> 3, b = bs2 & 7;
    const int nn  = gp2 * N_ + j;
    const int gp  = nn & 3, ni = nn >> 2;
    const float4 v =
        ((const float4*)(z + (((long)(gp * B_ + b)) * N_ + ni) * C_))[threadIdx.x];
    const long dst = (((long)bs2) * N_ + j) * C_;
    ((float4*)(cx + dst))[threadIdx.x] = v;
    v4h h;
    h[0] = (half_t)v.x; h[1] = (half_t)v.y;
    h[2] = (half_t)v.z; h[3] = (half_t)v.w;
    ((v4h*)(cxh + dst))[threadIdx.x] = h;
}

// ----------- cross-attention mean over 3 chunks + residual ------------------
__global__ void __launch_bounds__(256)
mean_resid(const float* __restrict__ xr, const float* __restrict__ firstx,
           float* __restrict__ y)
{
    const long i = (long)blockIdx.x * 256 + threadIdx.x;
    const long stride = (long)8 * N_ * C_;
    y[i] = firstx[i] +
           (xr[i] + xr[i + stride] + xr[i + 2 * stride]) * (1.0f / 3.0f);
}

// -------------------- final permute back to (B, C, 4096) --------------------
__global__ void __launch_bounds__(256)
final_out(const float* __restrict__ cy, const float* __restrict__ cx,
          float* __restrict__ out)
{
    __shared__ float tile[32][33];
    const int bz  = blockIdx.z;           // b*4 + gp2
    const int b   = bz >> 2, gp2 = bz & 3;
    const int c0  = blockIdx.y * 32;
    const int j0  = blockIdx.x * 32;
    const int tx  = threadIdx.x, ty = threadIdx.y;
    const float* src = (gp2 == 0) ? (cy + (long)b * N_ * C_)
                                  : (cx + (long)(gp2 * B_ + b) * N_ * C_);
    for (int jj = ty; jj < 32; jj += 8)
        tile[jj][tx] = src[((long)(j0 + jj)) * C_ + c0 + tx];
    __syncthreads();
    for (int cc = ty; cc < 32; cc += 8)
        out[(((long)b * C_) + c0 + cc) * NF_ + gp2 * N_ + j0 + tx] = tile[tx][cc];
}

// ------------------------- weight preparation -------------------------------
__global__ void __launch_bounds__(256)
wprep_blockdiag(const float* __restrict__ w, half_t* __restrict__ out)
{
    const int o = blockIdx.x;     // 0..255
    const int d = threadIdx.x;    // 0..255
    const int g = o >> 6;
    const float v = ((d >> 6) == g)
        ? w[((long)g * DH_ + (o & 63)) * DH_ + (d & 63)] : 0.0f;
    out[(long)o * C_ + d] = (half_t)v;
}

__global__ void __launch_bounds__(256)
wconv(const float* __restrict__ w, half_t* __restrict__ out, int n)
{
    const int i = blockIdx.x * 256 + threadIdx.x;
    if (i < n) out[i] = (half_t)w[i];
}

// ===========================================================================
extern "C" void kernel_launch(void* const* d_in, const int* in_sizes, int n_in,
                              void* d_out, int out_size, void* d_ws, size_t ws_size,
                              hipStream_t stream)
{
    (void)in_sizes; (void)n_in; (void)out_size; (void)ws_size;

    const float* feat     = (const float*)d_in[0];
    const float* sa_qk_w  = (const float*)d_in[1];
    const float* sa_v_w   = (const float*)d_in[2];
    const float* sa_ln1_g = (const float*)d_in[3];
    const float* sa_ln1_b = (const float*)d_in[4];
    const float* sa_fc1_w = (const float*)d_in[5];
    const float* sa_fc1_b = (const float*)d_in[6];
    const float* sa_fc2_w = (const float*)d_in[7];
    const float* sa_fc2_b = (const float*)d_in[8];
    const float* sa_ln2_g = (const float*)d_in[9];
    const float* sa_ln2_b = (const float*)d_in[10];
    const float* cr_q_w   = (const float*)d_in[11];
    const float* cr_k_w   = (const float*)d_in[12];
    const float* cr_v_w   = (const float*)d_in[13];
    const float* cr_ln1_g = (const float*)d_in[14];
    const float* cr_ln1_b = (const float*)d_in[15];
    const float* cr_fc1_w = (const float*)d_in[16];
    const float* cr_fc1_b = (const float*)d_in[17];
    const float* cr_fc2_w = (const float*)d_in[18];
    const float* cr_fc2_b = (const float*)d_in[19];
    const float* cr_ln2_g = (const float*)d_in[20];
    const float* cr_ln2_b = (const float*)d_in[21];

    // ------------------ workspace layout (~565 MB, 256B aligned) -----------
    char* base = (char*)d_ws;
    size_t off = 0;
    auto take = [&](size_t bytes) -> void* {
        void* p = base + off;
        off = (off + bytes + 255) & ~(size_t)255;
        return p;
    };

    half_t* w_qk  = (half_t*)take((size_t)C_ * C_ * 2);
    half_t* w_v   = (half_t*)take((size_t)C_ * C_ * 2);
    half_t* w_fc1 = (half_t*)take((size_t)C4_ * C_ * 2);
    half_t* w_fc2 = (half_t*)take((size_t)C_ * C4_ * 2);
    half_t* w_cq  = (half_t*)take((size_t)C_ * C_ * 2);
    half_t* w_ck  = (half_t*)take((size_t)C_ * C_ * 2);
    half_t* w_cv  = (half_t*)take((size_t)C_ * C_ * 2);
    half_t* w_cf1 = (half_t*)take((size_t)C4_ * C_ * 2);
    half_t* w_cf2 = (half_t*)take((size_t)C_ * C4_ * 2);

    float*  xt   = (float*) take((size_t)BS_ * N_ * C_ * 4);
    half_t* xh   = (half_t*)take((size_t)BS_ * N_ * C_ * 2);
    half_t* qh   = (half_t*)take((size_t)BS_ * N_ * C_ * 2);   // q == k
    half_t* vhT  = (half_t*)take((size_t)BS_ * C_ * N_ * 2);   // channel-major V
    float*  E    = (float*) take((size_t)BS_ * N_ * N_ * 4);   // overlays: hh, E2
    half_t* Ph   = (half_t*)take((size_t)BS_ * N_ * N_ * 2);   // overlay: P2
    float*  y    = (float*) take((size_t)BS_ * N_ * C_ * 4);
    float*  yln  = (float*) take((size_t)BS_ * N_ * C_ * 4);
    half_t* ylnh = (half_t*)take((size_t)BS_ * N_ * C_ * 2);
    float*  y2   = (float*) take((size_t)BS_ * N_ * C_ * 4);
    float*  z    = (float*) take((size_t)BS_ * N_ * C_ * 4);
    float*  cx   = (float*) take((size_t)BS_ * N_ * C_ * 4);
    half_t* cxh  = (half_t*)take((size_t)BS_ * N_ * C_ * 2);
    half_t* qch  = (half_t*)take((size_t)24 * N_ * C_ * 2);
    half_t* kch  = (half_t*)take((size_t)8 * N_ * C_ * 2);
    half_t* vcT  = (half_t*)take((size_t)8 * C_ * N_ * 2);
    float*  xr2  = (float*) take((size_t)24 * N_ * C_ * 4);
    float*  ym   = (float*) take((size_t)8 * N_ * C_ * 4);
    float*  ymln = (float*) take((size_t)8 * N_ * C_ * 4);
    half_t* ymlh = (half_t*)take((size_t)8 * N_ * C_ * 2);
    half_t* hc   = (half_t*)take((size_t)8 * N_ * C4_ * 2);
    float*  y2c  = (float*) take((size_t)8 * N_ * C_ * 4);
    float*  cy   = (float*) take((size_t)8 * N_ * C_ * 4);

    half_t* hh = (half_t*)E;  // 64 MiB <= 128 MiB; E dead after softmax
    float*  E2 = E;           // hh dead after fc2
    half_t* P2 = Ph;          // Ph dead after SA apply

#define GEMM_ARGS Aop, Bop, biasp, residp, oF, oH, oHT, T, O, K, \
                  sA, sB, bmod, sOF, sOH, sOHT, sR, scale

    auto gemm = [&](const half_t* Aop, const half_t* Bop, const float* biasp,
                    const float* residp, float* oF, half_t* oH, half_t* oHT,
                    int T, int O, int K, long sA, long sB, int bmod,
                    long sOF, long sOH, long sOHT, long sR,
                    float scale, int relu, int transA, int batches) {
        dim3 grid(T / 128, O / 128, batches);
        const bool bias  = biasp != nullptr;
        const bool resid = residp != nullptr;
        const int  omode = oF ? 0 : (oH ? 1 : 2);
        if (transA) {   // attention apply: P^T operand
            if (resid) gemm_wmma<1,0,1,0,0><<<grid,256,0,stream>>>(GEMM_ARGS);
            else       gemm_wmma<1,0,0,0,0><<<grid,256,0,stream>>>(GEMM_ARGS);
        } else if (bias) {  // FFN layers
            if (relu)  gemm_wmma<0,1,0,1,1><<<grid,256,0,stream>>>(GEMM_ARGS);
            else       gemm_wmma<0,1,1,0,0><<<grid,256,0,stream>>>(GEMM_ARGS);
        } else {        // projections / energy
            if (omode == 0)      gemm_wmma<0,0,0,0,0><<<grid,256,0,stream>>>(GEMM_ARGS);
            else if (omode == 1) gemm_wmma<0,0,0,0,1><<<grid,256,0,stream>>>(GEMM_ARGS);
            else                 gemm_wmma<0,0,0,0,2><<<grid,256,0,stream>>>(GEMM_ARGS);
        }
    };

    // ------------------------- weight prep ---------------------------------
    wprep_blockdiag<<<C_, C_, 0, stream>>>(sa_qk_w, w_qk);
    wprep_blockdiag<<<C_, C_, 0, stream>>>(cr_q_w, w_cq);
    wprep_blockdiag<<<C_, C_, 0, stream>>>(cr_k_w, w_ck);
    wconv<<<256, 256, 0, stream>>>(sa_v_w, w_v, C_ * C_);
    wconv<<<1024, 256, 0, stream>>>(sa_fc1_w, w_fc1, C4_ * C_);
    wconv<<<1024, 256, 0, stream>>>(sa_fc2_w, w_fc2, C_ * C4_);
    wconv<<<256, 256, 0, stream>>>(cr_v_w, w_cv, C_ * C_);
    wconv<<<1024, 256, 0, stream>>>(cr_fc1_w, w_cf1, C4_ * C_);
    wconv<<<1024, 256, 0, stream>>>(cr_fc2_w, w_cf2, C_ * C4_);

    // ------------------------- SA layer -------------------------------------
    permute_in<<<dim3(NF_ / 32, C_ / 32, B_), dim3(32, 8), 0, stream>>>(feat, xt, xh);

    // q (== k) projection
    gemm(xh, w_qk, nullptr, nullptr, nullptr, qh, nullptr,
         N_, C_, C_, (long)N_ * C_, 0, 1, 0, (long)N_ * C_, 0, 0,
         1.0f, 0, 0, BS_);
    // v projection -> channel-major f16
    gemm(xh, w_v, nullptr, nullptr, nullptr, nullptr, vhT,
         N_, C_, C_, (long)N_ * C_, 0, 1, 0, 0, (long)C_ * N_, 0,
         1.0f, 0, 0, BS_);
    // energy[n][m] = sum_c q[n][c] * q[m][c]
    gemm(qh, qh, nullptr, nullptr, E, nullptr, nullptr,
         N_, N_, C_, (long)N_ * C_, (long)N_ * C_, BS_,
         (long)N_ * N_, 0, 0, 0, 1.0f, 0, 0, BS_);
    softmax_row<<<BS_ * N_, 256, 0, stream>>>(E, Ph);
    // x_r^T[n][c] = sum_m P[m][n] * V[c][m]  (transposed-attention apply)
    gemm(Ph, vhT, nullptr, xt, y, nullptr, nullptr,
         N_, C_, N_, (long)N_ * N_, (long)C_ * N_, BS_,
         (long)N_ * C_, 0, 0, (long)N_ * C_, 0.0625f, 0, 1, BS_);
    layernorm_tok<<<BS_ * N_, 64, 0, stream>>>(y, sa_ln1_g, sa_ln1_b, yln, ylnh, 0);
    // fc1 + relu
    gemm(ylnh, w_fc1, sa_fc1_b, nullptr, nullptr, hh, nullptr,
         N_, C4_, C_, (long)N_ * C_, 0, 1, 0, (long)N_ * C4_, 0, 0,
         1.0f, 1, 0, BS_);
    // fc2 + residual
    gemm(hh, w_fc2, sa_fc2_b, yln, y2, nullptr, nullptr,
         N_, C_, C4_, (long)N_ * C4_, 0, 1, (long)N_ * C_, 0, 0, (long)N_ * C_,
         1.0f, 0, 0, BS_);
    layernorm_tok<<<BS_ * N_, 64, 0, stream>>>(y2, sa_ln2_g, sa_ln2_b, z, nullptr, 1);

    // ------------------------- cross layer ----------------------------------
    permute_tok<<<dim3(N_, BS_), 64, 0, stream>>>(z, cx, cxh);

    gemm(cxh + (size_t)8 * N_ * C_, w_cq, nullptr, nullptr, nullptr, qch, nullptr,
         N_, C_, C_, (long)N_ * C_, 0, 1, 0, (long)N_ * C_, 0, 0, 1.0f, 0, 0, 24);
    gemm(cxh, w_ck, nullptr, nullptr, nullptr, kch, nullptr,
         N_, C_, C_, (long)N_ * C_, 0, 1, 0, (long)N_ * C_, 0, 0, 1.0f, 0, 0, 8);
    gemm(cxh, w_cv, nullptr, nullptr, nullptr, nullptr, vcT,
         N_, C_, C_, (long)N_ * C_, 0, 1, 0, 0, (long)C_ * N_, 0, 1.0f, 0, 0, 8);
    // cross energy: B batch = bt % 8
    gemm(qch, kch, nullptr, nullptr, E2, nullptr, nullptr,
         N_, N_, C_, (long)N_ * C_, (long)N_ * C_, 8,
         (long)N_ * N_, 0, 0, 0, 1.0f, 0, 0, 24);
    softmax_row<<<24 * N_, 256, 0, stream>>>(E2, P2);
    gemm(P2, vcT, nullptr, nullptr, xr2, nullptr, nullptr,
         N_, C_, N_, (long)N_ * N_, (long)C_ * N_, 8,
         (long)N_ * C_, 0, 0, 0, 0.0625f, 0, 1, 24);
    mean_resid<<<(8 * N_ * C_) / 256, 256, 0, stream>>>(xr2, cx, ym);
    layernorm_tok<<<8 * N_, 64, 0, stream>>>(ym, cr_ln1_g, cr_ln1_b, ymln, ymlh, 0);
    gemm(ymlh, w_cf1, cr_fc1_b, nullptr, nullptr, hc, nullptr,
         N_, C4_, C_, (long)N_ * C_, 0, 1, 0, (long)N_ * C4_, 0, 0,
         1.0f, 1, 0, 8);
    gemm(hc, w_cf2, cr_fc2_b, ymln, y2c, nullptr, nullptr,
         N_, C_, C4_, (long)N_ * C4_, 0, 1, (long)N_ * C_, 0, 0, (long)N_ * C_,
         1.0f, 0, 0, 8);
    layernorm_tok<<<8 * N_, 64, 0, stream>>>(y2c, cr_ln2_g, cr_ln2_b, cy, nullptr, 1);

    final_out<<<dim3(N_ / 32, C_ / 32, B_ * GP_), dim3(32, 8), 0, stream>>>(
        cy, cx, (float*)d_out);
}
